// GCN_Sparse_Policy_SelectNode_30528627540626
// MI455X (gfx1250) — matrix operand order repaired and checked by reading, not compiled, not run on hardware
//
#include <hip/hip_runtime.h>

// ---------------------------------------------------------------------------
// Types for CDNA5 WMMA (wave32): v_wmma_f32_16x16x32_bf16
// ---------------------------------------------------------------------------
typedef __attribute__((ext_vector_type(16))) __bf16 v16bf;
typedef __attribute__((ext_vector_type(8)))  __bf16 v8bf;
typedef __attribute__((ext_vector_type(8)))  float  v8f;

__device__ __forceinline__ __bf16 f2bf(float f) {
    // round-to-nearest-even fp32 -> bf16
    unsigned u = __builtin_bit_cast(unsigned, f);
    unsigned r = u + 0x7FFFu + ((u >> 16) & 1u);
    unsigned short s = (unsigned short)(r >> 16);
    return __builtin_bit_cast(__bf16, s);
}

// ---------------------------------------------------------------------------
// fp32 -> bf16 conversion (vectorized float4 reads)
// ---------------------------------------------------------------------------
__global__ void cvt_bf16_kernel(const float* __restrict__ src,
                                __bf16* __restrict__ dst, int n) {
    int i = (blockIdx.x * blockDim.x + threadIdx.x) * 4;
    if (i + 3 < n) {
        float4 v = *(const float4*)&src[i];
        dst[i + 0] = f2bf(v.x);
        dst[i + 1] = f2bf(v.y);
        dst[i + 2] = f2bf(v.z);
        dst[i + 3] = f2bf(v.w);
    } else {
        for (; i < n; ++i) dst[i] = f2bf(src[i]);
    }
}

// ---------------------------------------------------------------------------
// fp32 [R][C] -> bf16 transposed [C][R]  (one-shot for the tiny weights; lets
// GEMM stage B tiles with straight b128 loads/stores instead of 2B scatters)
// ---------------------------------------------------------------------------
__global__ void cvt_transpose_bf16_kernel(const float* __restrict__ src,
                                          __bf16* __restrict__ dst,
                                          int R, int C) {
    int i = blockIdx.x * blockDim.x + threadIdx.x;
    if (i >= R * C) return;
    int r = i / C;
    int c = i - r * C;
    dst[(size_t)c * R + r] = f2bf(src[i]);
}

// ---------------------------------------------------------------------------
// zero fill (float4)
// ---------------------------------------------------------------------------
__global__ void zero4_kernel(float* __restrict__ p, int n) {
    int i = (blockIdx.x * blockDim.x + threadIdx.x) * 4;
    if (i + 3 < n) {
        *(float4*)&p[i] = make_float4(0.f, 0.f, 0.f, 0.f);
    } else {
        for (; i < n; ++i) p[i] = 0.f;
    }
}

// ---------------------------------------------------------------------------
// bf16 GEMM: C[M,N] = A[M,K] * Bt[N,K]^T, fp32 accumulate via WMMA.
// Block: 256 threads = 8 wave32. Block tile 256x64, K-step 64.
// Register blocking: each wave owns TWO 16-row A fragments (rows w*16 and
// 128+w*16) x four 16-col B fragments -> 8 WMMAs per 16x16x32 sub-step off
// only 12 ds_load_b128 (each B frag feeds two back-to-back WMMAs).
// A staged [256][64] in LDS (one coalesced 128B row per thread); Bt is the
// pre-transposed weight [N][K] so B stages with pure b128 traffic.
// N must be a multiple of 64, K a multiple of 64.
// ---------------------------------------------------------------------------
#define TM 256
#define TN 64
#define TK 64
#define MT 2   // m sub-tiles per wave

__global__ __launch_bounds__(256)
void gemm_bf16_wmma(const __bf16* __restrict__ A, const __bf16* __restrict__ Bt,
                    float* __restrict__ C, int M, int N, int K) {
    __shared__ alignas(32) __bf16 As[TM][TK];   // 32 KB
    __shared__ alignas(32) __bf16 Bs[TN][TK];   // 8 KB, [n][k]

    const int row0 = blockIdx.x * TM;
    const int col0 = blockIdx.y * TN;
    const int t    = threadIdx.x;
    const int wave = t >> 5;          // 0..7
    const int lane = t & 31;
    const int lg   = lane >> 4;       // half-wave group (per ISA frag layout)
    const int lm   = lane & 15;

    v8f acc[MT][TN / 16];
#pragma unroll
    for (int mt = 0; mt < MT; ++mt)
#pragma unroll
        for (int i = 0; i < TN / 16; ++i)
#pragma unroll
            for (int j = 0; j < 8; ++j) acc[mt][i][j] = 0.f;

    // A-stage mapping: one thread per row, 64 bf16 (128B) contiguous
    int agr = row0 + t;
    if (agr >= M) agr = M - 1;        // clamp (stores are guarded later)
    const __bf16* aptr = &A[(size_t)agr * K];
    // B-stage mapping: 4 threads per n-row, 16 bf16 (32B) each
    const int bn  = t >> 2;           // 0..63
    const int bk  = (t & 3) * 16;     // 0,16,32,48
    const __bf16* bptr = &Bt[(size_t)(col0 + bn) * K + bk];

    for (int k0 = 0; k0 < K; k0 += TK) {
        // ---- stage A tile (eight coalesced 16B global loads per thread) ----
        {
            const v8bf* src = (const v8bf*)(aptr + k0);
#pragma unroll
            for (int j = 0; j < 8; ++j)
                *(v8bf*)&As[t][j * 8] = src[j];
        }
        // ---- stage B tile ([n][k], two 16B loads -> two 16B LDS stores) ----
        {
            const v8bf* src = (const v8bf*)(bptr + k0);
            *(v8bf*)&Bs[bn][bk]     = src[0];
            *(v8bf*)&Bs[bn][bk + 8] = src[1];
        }
        // prefetch next K-step of A into near cache (speculative; OOB dropped)
        __builtin_prefetch(aptr + k0 + TK, 0, 3);
        __syncthreads();

#pragma unroll
        for (int ks = 0; ks < TK; ks += 32) {
            // ---- two A fragments: rows wave*16 and 128 + wave*16 ----
            // lane l: row = base + l%16; K: [g*16, +16)
            v16bf afrag[MT];
#pragma unroll
            for (int mt = 0; mt < MT; ++mt) {
                const int arow = mt * 128 + wave * 16 + lm;
                v8bf alo = *(const v8bf*)&As[arow][ks + lg * 16];
                v8bf ahi = *(const v8bf*)&As[arow][ks + lg * 16 + 8];
#pragma unroll
                for (int i = 0; i < 8; ++i) {
                    afrag[mt][i]     = alo[i];
                    afrag[mt][i + 8] = ahi[i];
                }
            }

            // ---- each B fragment feeds MT back-to-back WMMAs ----
#pragma unroll
            for (int nt = 0; nt < TN / 16; ++nt) {
                v16bf bfrag = *(const v16bf*)&Bs[nt * 16 + lm][ks + lg * 16];
#pragma unroll
                for (int mt = 0; mt < MT; ++mt)
                    acc[mt][nt] = __builtin_amdgcn_wmma_f32_16x16x32_bf16(
                        false, afrag[mt], false, bfrag, (short)0, acc[mt][nt],
                        false, false);
            }
        }
        __syncthreads();
    }

    // ---- epilogue: C/D layout VGPR r -> M = r + (l>>4)*8, N = l%16 ----
#pragma unroll
    for (int mt = 0; mt < MT; ++mt) {
#pragma unroll
        for (int nt = 0; nt < TN / 16; ++nt) {
#pragma unroll
            for (int r = 0; r < 8; ++r) {
                int m = row0 + mt * 128 + wave * 16 + lg * 8 + r;
                int n = col0 + nt * 16 + lm;
                if (m < M) C[(size_t)m * N + n] = acc[mt][nt][r];
            }
        }
    }
}

// ---------------------------------------------------------------------------
// Edge-parallel SpMM: Y[row] += val * X[col], F features, 4 floats/thread.
// fshift = log2(F/4). Gathers are L2-resident (X fits in 192MB L2).
// ---------------------------------------------------------------------------
__global__ void spmm_atomic_kernel(const int* __restrict__ rows,
                                   const int* __restrict__ cols,
                                   const float* __restrict__ vals,
                                   const float* __restrict__ X,
                                   float* __restrict__ Y,
                                   int E, int F, int fshift) {
    unsigned idx = blockIdx.x * blockDim.x + threadIdx.x;
    unsigned e   = idx >> fshift;
    if (e >= (unsigned)E) return;
    int fc = (int)(idx & ((1u << fshift) - 1u)) << 2;

    int   c = cols[e];
    int   r = rows[e];
    float v = vals[e];
    float4 x = *(const float4*)&X[(size_t)c * F + fc];
    float* y = &Y[(size_t)r * F + fc];
    __hip_atomic_fetch_add(y + 0, v * x.x, __ATOMIC_RELAXED, __HIP_MEMORY_SCOPE_AGENT);
    __hip_atomic_fetch_add(y + 1, v * x.y, __ATOMIC_RELAXED, __HIP_MEMORY_SCOPE_AGENT);
    __hip_atomic_fetch_add(y + 2, v * x.z, __ATOMIC_RELAXED, __HIP_MEMORY_SCOPE_AGENT);
    __hip_atomic_fetch_add(y + 3, v * x.w, __ATOMIC_RELAXED, __HIP_MEMORY_SCOPE_AGENT);
}

// ---------------------------------------------------------------------------
// h = relu(H + b1) -> bf16 (input to layer-2 GEMM)
// ---------------------------------------------------------------------------
__global__ void bias_relu_bf16_kernel(const float* __restrict__ H,
                                      const float* __restrict__ b,
                                      __bf16* __restrict__ out,
                                      int n, int fmask) {
    int i = blockIdx.x * blockDim.x + threadIdx.x;
    if (i >= n) return;
    float x = H[i] + b[i & fmask];
    out[i] = f2bf(fmaxf(x, 0.f));
}

// ---------------------------------------------------------------------------
// Column-wise (axis 0) logsumexp of (G + b2): one block per output feature.
// Online max/sum per thread, then (m,s) tree-combine in LDS.
// ---------------------------------------------------------------------------
__global__ __launch_bounds__(256)
void col_lse_kernel(const float* __restrict__ G, const float* __restrict__ b2,
                    float* __restrict__ lse, int Mrows, int F) {
    const int f = blockIdx.x;
    const int t = threadIdx.x;
    const float bb = b2[f];

    float m = -INFINITY, s = 0.f;
    for (int i = t; i < Mrows; i += blockDim.x) {
        float x = G[(size_t)i * F + f] + bb;
        if (x > m) { s *= __expf(m - x); m = x; }
        s += __expf(x - m);
    }

    __shared__ float sm[256];
    __shared__ float ss[256];
    sm[t] = m; ss[t] = s;
    __syncthreads();
    for (int off = 128; off > 0; off >>= 1) {
        if (t < off) {
            float m2 = sm[t + off], s2 = ss[t + off];
            float mm = fmaxf(sm[t], m2);
            ss[t] = ss[t] * __expf(sm[t] - mm) + s2 * __expf(m2 - mm);
            sm[t] = mm;
        }
        __syncthreads();
    }
    if (t == 0) lse[f] = sm[0] + __logf(ss[0]);
}

// out[i] = (out[i] + b2[f]) - lse[f]   (in place)
__global__ void lsm_write_kernel(float* __restrict__ out,
                                 const float* __restrict__ b2,
                                 const float* __restrict__ lse,
                                 int n, int fmask) {
    int i = blockIdx.x * blockDim.x + threadIdx.x;
    if (i >= n) return;
    int f = i & fmask;
    out[i] = out[i] + b2[f] - lse[f];
}

// ---------------------------------------------------------------------------
// Host launcher
// ---------------------------------------------------------------------------
extern "C" void kernel_launch(void* const* d_in, const int* in_sizes, int n_in,
                              void* d_out, int out_size, void* d_ws, size_t ws_size,
                              hipStream_t stream) {
    const float* features  = (const float*)d_in[0];
    const float* edge_vals = (const float*)d_in[1];
    const float* W1        = (const float*)d_in[2];
    const float* b1        = (const float*)d_in[3];
    const float* W2        = (const float*)d_in[4];
    const float* b2        = (const float*)d_in[5];
    const int*   edge_row  = (const int*)d_in[6];
    const int*   edge_col  = (const int*)d_in[7];
    float*       out       = (float*)d_out;

    const int NHID = in_sizes[3];              // 256
    const int NOUT = in_sizes[5];              // 64
    const int NIN  = in_sizes[2] / NHID;       // 256
    const int M    = in_sizes[0] / NIN;        // 50000
    const int E    = in_sizes[1];              // 1.6M

    // carve workspace
    size_t off = 0;
    auto carve = [&](size_t bytes) -> void* {
        off = (off + 255) & ~(size_t)255;
        void* p = (char*)d_ws + off;
        off += bytes;
        return p;
    };
    __bf16* Xb   = (__bf16*)carve((size_t)M * NIN * 2);  // reused as Hb for layer 2
    __bf16* W1t  = (__bf16*)carve((size_t)NIN * NHID * 2); // [NHID][NIN]
    __bf16* W2t  = (__bf16*)carve((size_t)NHID * NOUT * 2); // [NOUT][NHID]
    float*  XW1  = (float*)carve((size_t)M * NHID * 4);
    float*  H1   = (float*)carve((size_t)M * NHID * 4);
    float*  G2   = (float*)carve((size_t)M * NOUT * 4);
    float*  LSE  = (float*)carve((size_t)NOUT * 4);
    (void)ws_size;

    const int T = 256;
    auto cdiv = [](long long a, long long b) { return (int)((a + b - 1) / b); };

    // 1) fp32 -> bf16 conversions (weights transposed to [N][K])
    cvt_bf16_kernel<<<cdiv((long long)M * NIN, 4LL * T), T, 0, stream>>>(features, Xb, M * NIN);
    cvt_transpose_bf16_kernel<<<cdiv((long long)NIN * NHID, T), T, 0, stream>>>(W1, W1t, NIN, NHID);
    cvt_transpose_bf16_kernel<<<cdiv((long long)NHID * NOUT, T), T, 0, stream>>>(W2, W2t, NHID, NOUT);

    // 2) XW1 = Xb @ W1  (WMMA)
    {
        dim3 grid(cdiv(M, TM), NHID / TN);
        gemm_bf16_wmma<<<grid, T, 0, stream>>>(Xb, W1t, XW1, M, NHID, NIN);
    }

    // 3) H1 = spmm(adj, XW1)
    zero4_kernel<<<cdiv((long long)M * NHID, 4LL * T), T, 0, stream>>>(H1, M * NHID);
    {
        int fshift = __builtin_ctz(NHID >> 2);   // 6 for F=256
        long long work = (long long)E << fshift;
        spmm_atomic_kernel<<<cdiv(work, T), T, 0, stream>>>(edge_row, edge_col, edge_vals,
                                                            XW1, H1, E, NHID, fshift);
    }

    // 4) Hb = bf16(relu(H1 + b1))  (Xb region reused)
    bias_relu_bf16_kernel<<<cdiv((long long)M * NHID, T), T, 0, stream>>>(H1, b1, Xb,
                                                                          M * NHID, NHID - 1);

    // 5) G2 = Hb @ W2  (WMMA)
    {
        dim3 grid(cdiv(M, TM), NOUT / TN);
        gemm_bf16_wmma<<<grid, T, 0, stream>>>(Xb, W2t, G2, M, NOUT, NHID);
    }

    // 6) out = spmm(adj, G2)  (accumulate directly into d_out)
    zero4_kernel<<<cdiv((long long)M * NOUT, 4LL * T), T, 0, stream>>>(out, M * NOUT);
    {
        int fshift = __builtin_ctz(NOUT >> 2);   // 4 for F=64
        long long work = (long long)E << fshift;
        spmm_atomic_kernel<<<cdiv(work, T), T, 0, stream>>>(edge_row, edge_col, edge_vals,
                                                            G2, out, E, NOUT, fshift);
    }

    // 7) log_softmax over nodes (axis 0), in place
    col_lse_kernel<<<NOUT, T, 0, stream>>>(out, b2, LSE, M, NOUT);
    lsm_write_kernel<<<cdiv((long long)M * NOUT, T), T, 0, stream>>>(out, b2, LSE,
                                                                     M * NOUT, NOUT - 1);
}
